// NoiseLayer_75771813037095
// MI455X (gfx1250) — compile-verified, matrix operand order checked by reading, not run on hardware
//
#include <hip/hip_runtime.h>
#include <hip/hip_bf16.h>

typedef __attribute__((ext_vector_type(16))) _Float16 v16h;
typedef __attribute__((ext_vector_type(8)))  float    v8f;
typedef unsigned int u32x4 __attribute__((ext_vector_type(4)));
typedef int          i32x4 __attribute__((ext_vector_type(4)));
typedef int          i32x8 __attribute__((ext_vector_type(8)));

#define B_ROWS 16384
#define D_COLS 2048
#define NCLS   1000
#define ALPHA  0.1f

// workspace layout, in floats
#define WS_CNT    0                          // 1000 (padded 1024)
#define WS_S      1024                       // 1000*2048 (reused as noise table)
#define WS_SQ     (1024 + NCLS*D_COLS)       // 1000*2048
#define WS_NEWY   (1024 + 2*NCLS*D_COLS)     // 16384 ints
#define WS_FLOATS (WS_NEWY + B_ROWS)

// ---------------------------------------------------------------------------
// TDM: 2D strided tile load Global -> LDS via Tensor Data Mover.
// D# built per CDNA5 ISA 8.3/8.4: group0 = {count=1, lds_addr, global_addr,
// type=2}; group1 = {data_size=4B, tensor_dim0/1, tile_dim0/1, dim0_stride};
// groups 2/3 zero (2D tensor -> VADDR2/3 logically disabled).
// ---------------------------------------------------------------------------
#if __has_builtin(__builtin_amdgcn_tensor_load_to_lds)
#define HAVE_TDM 1
#warning "CDNA5 probe: __builtin_amdgcn_tensor_load_to_lds PRESENT (TDM path active)"
#if defined(__clang_major__) && (__clang_major__ >= 23)
#warning "CDNA5 probe: using 6-arg TDM builtin form (clang >= 23)"
#else
#warning "CDNA5 probe: using 5-arg TDM builtin form (clang <= 22)"
#endif
__device__ __forceinline__ void tdm_load_2d(unsigned lds_addr, const void* gptr,
                                            unsigned tile_w, unsigned tile_h,
                                            unsigned tensor_w, unsigned tensor_h,
                                            unsigned row_stride_elems) {
    const unsigned long long ga = (unsigned long long)(uintptr_t)gptr;
    u32x4 g0;
    g0[0] = 1u;                                             // count=1 (valid D#)
    g0[1] = lds_addr;                                       // lds_addr[31:0]
    g0[2] = (unsigned)(ga & 0xFFFFFFFFu);                   // global_addr[31:0]
    g0[3] = (unsigned)((ga >> 32) & 0x01FFFFFFu) | (2u << 30); // addr[56:32]|type=2
    i32x8 g1;
    g1[0] = (int)(2u << 16);                                // data_size=2 -> 4 bytes
    g1[1] = (int)((tensor_w & 0xFFFFu) << 16);              // tensor_dim0[15:0] @ [63:48]
    g1[2] = (int)(((tensor_w >> 16) & 0xFFFFu) | ((tensor_h & 0xFFFFu) << 16));
    g1[3] = (int)(((tensor_h >> 16) & 0xFFFFu) | ((tile_w & 0xFFFFu) << 16));
    g1[4] = (int)(tile_h & 0xFFFFu);                        // tile_dim1; tile_dim2=0
    g1[5] = (int)row_stride_elems;                          // tensor_dim0_stride[31:0]
    g1[6] = 0;                                              // stride hi / dim1_stride
    g1[7] = 0;
    i32x4 z4 = {0, 0, 0, 0};
#if defined(__clang_major__) && (__clang_major__ >= 23)
    i32x8 z8 = {0, 0, 0, 0, 0, 0, 0, 0};
    __builtin_amdgcn_tensor_load_to_lds(g0, g1, z4, z4, z8, 0);
#else
    __builtin_amdgcn_tensor_load_to_lds(g0, g1, z4, z4, 0);
#endif
}
#else
#define HAVE_TDM 0
#warning "CDNA5 probe: __builtin_amdgcn_tensor_load_to_lds ABSENT (fallback path)"
#endif

// ---------------------------------------------------------------------------
// Kernel 5 (defined FIRST so the TDM issue shows in the disasm snippet):
// out = 0.9*x + 0.1*noise[newY]. Each WG owns a 32-col slab and a 1024-row
// chunk; the 1000x32 noise slab (128 KB) is DMA'd into LDS by the TDM as one
// strided 2D tile, newY chunk staged too; x streams as float4 (128 B per row
// segment = one cache line).
// ---------------------------------------------------------------------------
__global__ void k_out(const float* __restrict__ x, const float* __restrict__ noise,
                      const int* __restrict__ newYi, float* __restrict__ out) {
    extern __shared__ float lds[];
    float* nslab = lds;                       // [1000][32]
    int*   nysh  = (int*)(lds + NCLS * 32);   // [1024]

    const int t       = threadIdx.x;          // 0..255
    const int colbase = blockIdx.x * 32;
    const int rowbase = blockIdx.y * 1024;

#if HAVE_TDM
    if (t < 32) {   // wave 0 issues the DMA and drains TENSORcnt
        tdm_load_2d((unsigned)(uintptr_t)nslab, noise + colbase,
                    /*tile_w=*/32, /*tile_h=*/NCLS,
                    /*tensor_w=*/D_COLS, /*tensor_h=*/NCLS,
                    /*row_stride=*/D_COLS);
        __builtin_amdgcn_s_wait_tensorcnt(0);
    }
#else
    for (int i = t * 4; i < NCLS * 32; i += 256 * 4) {
        *(float4*)(nslab + i) =
            *(const float4*)(noise + (size_t)(i >> 5) * D_COLS + colbase + (i & 31));
    }
#endif
    for (int i = t; i < 1024; i += 256) nysh[i] = newYi[rowbase + i];
    __syncthreads();

    const int q    = t & 7;                   // float4 column within slab
    const int rsub = t >> 3;                  // 32 rows in flight
    for (int rp = 0; rp < 1024; rp += 32) {
        const int rl  = rp + rsub;
        const int row = rowbase + rl;
        const size_t xoff = (size_t)row * D_COLS + colbase + q * 4;
        if (row + 32 < B_ROWS)
            __builtin_prefetch(x + xoff + (size_t)32 * D_COLS, 0, 1);
        const float4 xv = *(const float4*)(x + xoff);
        const int    ny = nysh[rl];
        const float4 nv = *(const float4*)(nslab + ny * 32 + q * 4);
        float4 o;
        o.x = (1.f - ALPHA) * xv.x + ALPHA * nv.x;
        o.y = (1.f - ALPHA) * xv.y + ALPHA * nv.y;
        o.z = (1.f - ALPHA) * xv.z + ALPHA * nv.z;
        o.w = (1.f - ALPHA) * xv.w + ALPHA * nv.w;
        *(float4*)(out + xoff) = o;
    }
}

// ---------------------------------------------------------------------------
// Kernel 1: per-class counts via WMMA one-hot x ones (exact in f16/f32).
// ---------------------------------------------------------------------------
__global__ void k_count_wmma(const int* __restrict__ y, float* __restrict__ cnt) {
    __shared__ int ysh[2048];
    const int lane    = threadIdx.x;          // 0..31, one wave32
    const int cb      = blockIdx.x;           // class block 0..62
    const int rowbase = blockIdx.y * 2048;    // row chunk

    for (int i = lane; i < 2048; i += 32) ysh[i] = y[rowbase + i];
    __syncthreads();

    const int m    = lane & 15;
    const int kofs = (lane >= 16) ? 8 : 0;
    const int cls  = cb * 16 + m;

    v16h bones;
#pragma unroll
    for (int e = 0; e < 16; ++e) bones[e] = (_Float16)1.0f;

    v8f acc = {};
    for (int t = 0; t < 64; ++t) {
        const int kt = t * 32;
        v16h a;
#pragma unroll
        for (int e = 0; e < 16; ++e) {
            const int r = e >> 1, h = e & 1;
            const int K = ((r < 4) ? (2 * r) : (16 + 2 * (r - 4))) + kofs + h;
            a[e] = (ysh[kt + K] == cls) ? (_Float16)1.0f : (_Float16)0.0f;
        }
        acc = __builtin_amdgcn_wmma_f32_16x16x32_f16(
            false, a, false, bones, (short)0, acc, false, false);
    }
    if (lane == 0 || lane == 16) {
        const int cbase = cb * 16 + ((lane == 16) ? 8 : 0);
#pragma unroll
        for (int r = 0; r < 8; ++r) {
            const int c = cbase + r;
            if (c < NCLS) atomicAdd(&cnt[c], acc[r]);
        }
    }
}

// ---------------------------------------------------------------------------
// Kernel 2: per-class sum / sum-of-squares, LDS-privatized 16-col slab.
// ---------------------------------------------------------------------------
__global__ void k_stats(const float* __restrict__ x, const int* __restrict__ y,
                        float* __restrict__ s, float* __restrict__ sq) {
    extern __shared__ float smem[];
    float* s_l  = smem;                // [1000*16]
    float* sq_l = smem + NCLS * 16;    // [1000*16]

    const int t = threadIdx.x;         // 0..255
    for (int i = t; i < NCLS * 16; i += 256) { s_l[i] = 0.f; sq_l[i] = 0.f; }
    __syncthreads();

    const int colbase = blockIdx.x * 16;
    const int rowbase = blockIdx.y * 2048;
    const int col  = t & 15;
    const int rsub = t >> 4;

    for (int rp = 0; rp < 2048; rp += 16) {
        const int row = rowbase + rp + rsub;
        if (row + 16 < B_ROWS)
            __builtin_prefetch(&x[(size_t)(row + 16) * D_COLS + colbase + col], 0, 1);
        const int   c = y[row];
        const float v = x[(size_t)row * D_COLS + colbase + col];
        atomicAdd(&s_l [c * 16 + col], v);
        atomicAdd(&sq_l[c * 16 + col], v * v);
    }
    __syncthreads();

    for (int i = t; i < NCLS * 16; i += 256) {
        const int c = i >> 4, cc = i & 15;
        atomicAdd(&s [(size_t)c * D_COLS + colbase + cc], s_l [i]);
        atomicAdd(&sq[(size_t)c * D_COLS + colbase + cc], sq_l[i]);
    }
}

// ---------------------------------------------------------------------------
// Kernel 3: deterministic permutation -> newY (int for gather, float to d_out).
// ---------------------------------------------------------------------------
__global__ void k_perm(const int* __restrict__ y, int* __restrict__ newYi,
                       float* __restrict__ newYf) {
    const int i = blockIdx.x * 256 + threadIdx.x;
    if (i >= B_ROWS) return;
    const int p = (i * 12869 + 7919) & (B_ROWS - 1);
    const int v = y[p];
    newYi[i] = v;
    newYf[i] = (float)v;
}

// ---------------------------------------------------------------------------
// Kernel 4: mean/std + Box-Muller gaussian; noise table in-place over s.
// ---------------------------------------------------------------------------
__device__ __forceinline__ unsigned hash_u32(unsigned v) {
    v ^= v >> 16; v *= 0x7feb352du;
    v ^= v >> 15; v *= 0x846ca68bu;
    v ^= v >> 16;
    return v;
}

__global__ void k_finalize(const float* __restrict__ cnt, float* __restrict__ s,
                           const float* __restrict__ sq) {
    const int idx = blockIdx.x * 256 + threadIdx.x;
    if (idx >= NCLS * D_COLS) return;
    const int   c   = idx / D_COLS;
    const float n   = cnt[c];
    const float sum = s[idx];
    const float ssq = sq[idx];
    float noise = 0.f;
    if (n >= 2.f) {
        const float mean = sum / n;
        const float var  = (ssq - n * mean * mean) / (n - 1.f);
        const float sd   = sqrtf(fmaxf(var, 0.f));
        const unsigned h1 = hash_u32((unsigned)idx * 2u + 0x9E3779B9u);
        const unsigned h2 = hash_u32((unsigned)idx * 2u + 0x85EBCA6Bu);
        const float u1 = fmaxf((float)(h1 >> 8) * (1.f / 16777216.f), 1e-7f);
        const float u2 = (float)(h2 >> 8) * (1.f / 16777216.f);
        const float g  = sqrtf(-2.f * logf(u1)) * cosf(6.2831853f * u2);
        noise = mean + sd * g;
    } else if (n >= 1.f) {
        noise = sum / n;
    }
    s[idx] = noise;
}

// ---------------------------------------------------------------------------
extern "C" void kernel_launch(void* const* d_in, const int* in_sizes, int n_in,
                              void* d_out, int out_size, void* d_ws, size_t ws_size,
                              hipStream_t stream) {
    (void)in_sizes; (void)n_in; (void)out_size;
    if (ws_size < (size_t)WS_FLOATS * sizeof(float)) return;

    const float* x = (const float*)d_in[0];
    const int*   y = (const int*)d_in[1];
    float* out = (float*)d_out;
    float* ws  = (float*)d_ws;

    float* cnt   = ws + WS_CNT;
    float* s     = ws + WS_S;     // later: noise table
    float* sq    = ws + WS_SQ;
    int*   newYi = (int*)(ws + WS_NEWY);
    float* newYf = out + (size_t)B_ROWS * D_COLS;

    hipMemsetAsync(d_ws, 0, (size_t)WS_NEWY * sizeof(float), stream);

    k_count_wmma<<<dim3(63, 8), 32, 0, stream>>>(y, cnt);
    k_stats<<<dim3(D_COLS / 16, 8), 256, 2 * NCLS * 16 * sizeof(float), stream>>>(x, y, s, sq);
    k_perm<<<(B_ROWS + 255) / 256, 256, 0, stream>>>(y, newYi, newYf);
    k_finalize<<<(NCLS * D_COLS + 255) / 256, 256, 0, stream>>>(cnt, s, sq);

    const size_t out_lds = (size_t)NCLS * 32 * sizeof(float) + 1024 * sizeof(int);
    k_out<<<dim3(D_COLS / 32, B_ROWS / 1024), 256, out_lds, stream>>>(x, s, newYi, out);
}
